// LSTMClassifier_412316860999
// MI455X (gfx1250) — compile-verified, matrix operand order checked by reading, not run on hardware
//
#include <hip/hip_runtime.h>
#include <hip/hip_bf16.h>

// ---------------------------------------------------------------------------
// Per-feature 2-layer LSTM (F=48 banks, H=256, B=32, T=128) + MLP head.
// Recurrent GEMMs run as bf16 WMMA 16x16x32 with f32 accumulation; weights are
// pre-converted once per launch into WMMA B-fragment order in d_ws so the hot
// loop does only contiguous b128 loads (L2-resident, ~75MB bf16 << 192MB L2).
// Grid: 96 blocks = 48 features x 2 batch-halves; 8 waves/block, each wave
// owns a 32-wide hidden slice and keeps its cell state in WMMA C-layout regs.
// Cell nonlinearities use gfx1250's native v_tanh_f32 when available.
// ---------------------------------------------------------------------------

typedef __attribute__((ext_vector_type(16))) __bf16 v16bf;
typedef __attribute__((ext_vector_type(8)))  float  v8f;

#define LB   128            // T
#define NF   48             // F
#define HH   256            // H
#define NB   32             // B
#define OUTW 40
#define CATW (NF*HH + HH)   // 12544

// frag layout: [f][mat(3)][ntile(64)][ktile(8)][lane(32)][16 bf16]
static constexpr size_t MAT_ELEMS  = (size_t)64 * 8 * 32 * 16;   // 262144 = 4H*H
static constexpr size_t FRAG_ELEMS = (size_t)NF * 3 * MAT_ELEMS; // 37,748,736
static constexpr size_t FRAG_BYTES = FRAG_ELEMS * 2;             // 75,497,472
static constexpr size_t CAT_BYTES  = (size_t)NB * CATW * 4;      // 1,605,632

// ---- fast nonlinearities: prefer gfx1250 native v_tanh_f32 ----------------
#if __has_builtin(__builtin_amdgcn_tanhf)
__device__ __forceinline__ float tanh_(float x) { return __builtin_amdgcn_tanhf(x); }
// sigmoid(x) = 0.5 * tanh(x/2) + 0.5  -> one trans op + one FMA
__device__ __forceinline__ float sigmoid_(float x) {
  return fmaf(0.5f, __builtin_amdgcn_tanhf(0.5f * x), 0.5f);
}
#else
__device__ __forceinline__ float tanh_(float x) { return tanhf(x); }
__device__ __forceinline__ float sigmoid_(float x) {
  return 1.0f / (1.0f + __expf(-x));
}
#endif

// 32B global B-fragment load (per-lane contiguous, 16B aligned)
__device__ __forceinline__ v16bf load_frag_g(const __bf16* p) {
  union { uint4 u[2]; v16bf v; } t;
  const uint4* q = reinterpret_cast<const uint4*>(p);
  t.u[0] = q[0];
  t.u[1] = q[1];
  return t.v;
}

// A-operand (16-bit A 16x32 layout): lane ln<16 rows M=ln; elements 0..7 hold
// K = kt*32 + hl*8 .. +7, elements 8..15 hold K = kt*32 + 16 + hl*8 .. +7.
__device__ __forceinline__ v16bf load_a_lds(const __bf16* hs, int ln, int hl, int kt) {
  union { uint4 u[2]; v16bf v; } t;
  const uint4* q0 = reinterpret_cast<const uint4*>(hs + ln * HH + kt * 32 + hl * 8);
  const uint4* q1 = reinterpret_cast<const uint4*>(hs + ln * HH + kt * 32 + 16 + hl * 8);
  t.u[0] = q0[0];
  t.u[1] = q1[0];
  return t.v;
}

// --------------------------- weight pre-pass -------------------------------
// One thread per (f, mat, ntile, ktile, lane): emit 16 bf16 in B-frag order.
// B[k][n] = W[f][n][k]  (einsum "bfh,fgh->bfg" => B-operand is W^T).
__global__ __launch_bounds__(256) void wconv_kernel(
    const float* __restrict__ Whh0, const float* __restrict__ Wih1,
    const float* __restrict__ Whh1, __bf16* __restrict__ frags) {
  size_t idx = (size_t)blockIdx.x * 256 + threadIdx.x;
  if (idx >= (size_t)NF * 3 * 64 * 8 * 32) return;
  int lane = (int)(idx & 31); size_t t = idx >> 5;
  int kt  = (int)(t & 7);  t >>= 3;
  int nti = (int)(t & 63); t >>= 6;
  int mat = (int)(t % 3);
  int f   = (int)(t / 3);
  const float* W  = (mat == 0) ? Whh0 : (mat == 1) ? Wih1 : Whh1;
  const float* Wf = W + (size_t)f * 4 * HH * HH;
  int n  = nti * 16 + (lane & 15);
  int hl = lane >> 4;
  const float* src = Wf + (size_t)n * HH + kt * 32;
  __bf16* out = frags + idx * 16;
#pragma unroll
  for (int e = 0; e < 16; ++e) {
    int kl = (e < 8) ? (hl * 8 + e) : (16 + hl * 8 + (e - 8));
    out[e] = (__bf16)src[kl];
  }
}

// ------------------------------ LSTM core ----------------------------------
__global__ __launch_bounds__(256) void lstm_kernel(
    const float* __restrict__ x, const float* __restrict__ Wih0,
    const float* __restrict__ b0, const float* __restrict__ b1,
    const __bf16* __restrict__ frags, float* __restrict__ cat) {
  const int f   = blockIdx.x >> 1;   // feature
  const int bh  = blockIdx.x & 1;    // batch half (rows bh*16 .. bh*16+15)
  const int tid = threadIdx.x;
  const int w    = tid >> 5;
  const int lane = tid & 31;
  const int ln   = lane & 15;
  const int hl   = lane >> 4;
  const int jb   = w * 32;           // this wave's hidden slice base

  __shared__ __bf16 h0s[16 * HH];
  __shared__ __bf16 h1s[16 * HH];
  __shared__ float  xs[16];

  for (int i = tid; i < 16 * HH; i += 256) { h0s[i] = (__bf16)0.0f; h1s[i] = (__bf16)0.0f; }

  // per-lane constants: biases and scalar-input weights at this lane's columns
  float bias0[4][2], bias1[4][2], wi0[4][2];
  const size_t fg = (size_t)f * 4 * HH;
#pragma unroll
  for (int g = 0; g < 4; ++g) {
#pragma unroll
    for (int nt = 0; nt < 2; ++nt) {
      int col = g * HH + jb + nt * 16 + ln;
      bias0[g][nt] = b0[fg + col];
      bias1[g][nt] = b1[fg + col];
      wi0[g][nt]   = Wih0[fg + col];
    }
  }

  const __bf16* fb0 = frags + (size_t)f * 3 * MAT_ELEMS;
  const __bf16* fb1 = fb0 + MAT_ELEMS;
  const __bf16* fb2 = fb0 + 2 * MAT_ELEMS;
  const int laneOff = lane * 16;

  v8f c0[2], c1[2];
#pragma unroll
  for (int nt = 0; nt < 2; ++nt) { c0[nt] = (v8f)0.0f; c1[nt] = (v8f)0.0f; }

  __syncthreads();

  for (int t = 0; t < LB; ++t) {
    if (tid < 16) xs[tid] = x[((size_t)(bh * 16 + tid) * LB + t) * NF + f];
    __syncthreads();

    float xrow[8];
#pragma unroll
    for (int r = 0; r < 8; ++r) xrow[r] = xs[hl * 8 + r];

    // ---- layer 0: g0 = x*Wih0 + b0 + h0_old @ Whh0^T ----
    v8f acc[4][2];
#pragma unroll
    for (int g = 0; g < 4; ++g)
#pragma unroll
      for (int nt = 0; nt < 2; ++nt)
#pragma unroll
        for (int r = 0; r < 8; ++r)
          acc[g][nt][r] = fmaf(xrow[r], wi0[g][nt], bias0[g][nt]);

#pragma unroll 4
    for (int kt = 0; kt < 8; ++kt) {
      v16bf a = load_a_lds(h0s, ln, hl, kt);
#pragma unroll
      for (int g = 0; g < 4; ++g) {
#pragma unroll
        for (int nt = 0; nt < 2; ++nt) {
          int nti = g * 16 + w * 2 + nt;
          v16bf bm = load_frag_g(fb0 + (size_t)((nti * 8 + kt) * 512 + laneOff));
          acc[g][nt] = __builtin_amdgcn_wmma_f32_16x16x32_bf16(
              false, a, false, bm, (short)0, acc[g][nt], false, false);
        }
      }
    }
    __syncthreads();  // all waves done reading h0_old

    // cell update layer 0 (element (r,lane) = row m=hl*8+r, col jb+nt*16+ln)
#pragma unroll
    for (int nt = 0; nt < 2; ++nt) {
#pragma unroll
      for (int r = 0; r < 8; ++r) {
        float ig  = sigmoid_(acc[0][nt][r]);
        float fgt = sigmoid_(acc[1][nt][r]);
        float gg  = tanh_(acc[2][nt][r]);
        float og  = sigmoid_(acc[3][nt][r]);
        float c   = fmaf(fgt, c0[nt][r], ig * gg);
        c0[nt][r] = c;
        float h   = og * tanh_(c);
        int m = hl * 8 + r;
        h0s[m * HH + jb + nt * 16 + ln] = (__bf16)h;
      }
    }
    __syncthreads();  // h0_new visible

    // ---- layer 1: g1 = b1 + h0_new @ Wih1^T + h1_old @ Whh1^T ----
#pragma unroll
    for (int g = 0; g < 4; ++g)
#pragma unroll
      for (int nt = 0; nt < 2; ++nt)
        acc[g][nt] = (v8f)bias1[g][nt];

#pragma unroll 4
    for (int kt = 0; kt < 8; ++kt) {
      v16bf a0 = load_a_lds(h0s, ln, hl, kt);
      v16bf a1 = load_a_lds(h1s, ln, hl, kt);
#pragma unroll
      for (int g = 0; g < 4; ++g) {
#pragma unroll
        for (int nt = 0; nt < 2; ++nt) {
          int nti = g * 16 + w * 2 + nt;
          v16bf bi  = load_frag_g(fb1 + (size_t)((nti * 8 + kt) * 512 + laneOff));
          v16bf bhf = load_frag_g(fb2 + (size_t)((nti * 8 + kt) * 512 + laneOff));
          acc[g][nt] = __builtin_amdgcn_wmma_f32_16x16x32_bf16(
              false, a0, false, bi, (short)0, acc[g][nt], false, false);
          acc[g][nt] = __builtin_amdgcn_wmma_f32_16x16x32_bf16(
              false, a1, false, bhf, (short)0, acc[g][nt], false, false);
        }
      }
    }
    __syncthreads();  // all waves done reading h1_old

    // cell update layer 1; final step also streams f32 h1 into cat buffer
#pragma unroll
    for (int nt = 0; nt < 2; ++nt) {
#pragma unroll
      for (int r = 0; r < 8; ++r) {
        float ig  = sigmoid_(acc[0][nt][r]);
        float fgt = sigmoid_(acc[1][nt][r]);
        float gg  = tanh_(acc[2][nt][r]);
        float og  = sigmoid_(acc[3][nt][r]);
        float c   = fmaf(fgt, c1[nt][r], ig * gg);
        c1[nt][r] = c;
        float h   = og * tanh_(c);
        int m = hl * 8 + r;
        h1s[m * HH + jb + nt * 16 + ln] = (__bf16)h;
        if (t == LB - 1) {
          cat[(size_t)(bh * 16 + m) * CATW + (size_t)f * HH + jb + nt * 16 + ln] = h;
        }
      }
    }
    __syncthreads();
  }
}

// ------------------------------- MLP head ----------------------------------
__global__ void head_spec_kernel(const float* __restrict__ spec,
                                 const float* __restrict__ Ws,
                                 const float* __restrict__ bs,
                                 float* __restrict__ cat) {
  int idx = blockIdx.x * 256 + threadIdx.x;
  if (idx >= NB * HH) return;
  int b = idx >> 8, j = idx & 255;
  float a = bs[j];
#pragma unroll
  for (int s = 0; s < 8; ++s) a = fmaf(spec[b * 8 + s], Ws[j * 8 + s], a);
  cat[(size_t)b * CATW + NF * HH + j] = a;
}

__global__ void head_fc1_kernel(const float* __restrict__ cat,
                                const float* __restrict__ Wp1,
                                const float* __restrict__ bp1,
                                float* __restrict__ hid) {
  int idx = blockIdx.x * 256 + threadIdx.x;
  if (idx >= NB * HH) return;
  int b = idx >> 8, j = idx & 255;
  const float4* c4 = reinterpret_cast<const float4*>(cat + (size_t)b * CATW);
  const float4* w4 = reinterpret_cast<const float4*>(Wp1 + (size_t)j * CATW);
  float a = bp1[j];
  for (int k = 0; k < CATW / 4; ++k) {
    float4 cv = c4[k], wv = w4[k];
    a = fmaf(cv.x, wv.x, a); a = fmaf(cv.y, wv.y, a);
    a = fmaf(cv.z, wv.z, a); a = fmaf(cv.w, wv.w, a);
  }
  hid[idx] = fmaxf(a, 0.0f);
}

__global__ void head_fc2_kernel(const float* __restrict__ hid,
                                const float* __restrict__ Wp2,
                                const float* __restrict__ bp2,
                                float* __restrict__ y) {
  int idx = blockIdx.x * 256 + threadIdx.x;
  if (idx >= NB * OUTW) return;
  int b = idx / OUTW, o = idx % OUTW;
  float a = bp2[o];
  for (int k = 0; k < HH; ++k) a = fmaf(hid[b * HH + k], Wp2[o * HH + k], a);
  y[idx] = a;
}

// ------------------------------- launcher ----------------------------------
extern "C" void kernel_launch(void* const* d_in, const int* in_sizes, int n_in,
                              void* d_out, int out_size, void* d_ws, size_t ws_size,
                              hipStream_t stream) {
  const float* x    = (const float*)d_in[0];
  const float* spec = (const float*)d_in[1];
  const float* Wih0 = (const float*)d_in[2];
  const float* Whh0 = (const float*)d_in[3];
  const float* b0   = (const float*)d_in[4];
  const float* Wih1 = (const float*)d_in[5];
  const float* Whh1 = (const float*)d_in[6];
  const float* b1   = (const float*)d_in[7];
  const float* Ws   = (const float*)d_in[8];
  const float* bs   = (const float*)d_in[9];
  const float* Wp1  = (const float*)d_in[10];
  const float* bp1  = (const float*)d_in[11];
  const float* Wp2  = (const float*)d_in[12];
  const float* bp2  = (const float*)d_in[13];

  __bf16* frags = (__bf16*)d_ws;
  float*  cat   = (float*)((char*)d_ws + FRAG_BYTES);
  float*  hid   = (float*)((char*)d_ws + FRAG_BYTES + CAT_BYTES);
  float*  y     = (float*)d_out;

  // 1) convert f32 weights -> bf16 WMMA B-fragments (one thread per frag-lane)
  {
    size_t total = (size_t)NF * 3 * 64 * 8 * 32;
    int blocks = (int)((total + 255) / 256);
    wconv_kernel<<<blocks, 256, 0, stream>>>(Whh0, Wih1, Whh1, frags);
  }
  // 2) recurrent core: 48 features x 2 batch-halves, 8 waves each
  lstm_kernel<<<NF * 2, 256, 0, stream>>>(x, Wih0, b0, b1, frags, cat);
  // 3) head
  head_spec_kernel<<<(NB * HH + 255) / 256, 256, 0, stream>>>(spec, Ws, bs, cat);
  head_fc1_kernel<<<(NB * HH + 255) / 256, 256, 0, stream>>>(cat, Wp1, bp1, hid);
  head_fc2_kernel<<<(NB * OUTW + 255) / 256, 256, 0, stream>>>(hid, Wp2, bp2, y);
}
